// aaTransformer_73263552135581
// MI455X (gfx1250) — matrix operand order repaired and checked by reading, not compiled, not run on hardware
//
#include <hip/hip_runtime.h>
#include <math.h>

// ============================================================================
// aaTransformer for MI455X (gfx1250, wave32, WMMA + TDM).
// All GEMMs -> v_wmma_f32_16x16x32_f16 (f32 accum).
//  - Aligned fast path: A tiles DMA'd to LDS by the Tensor Data Mover
//    (tensor_load_to_lds, double-buffered, overlapped with WMMA), B tiles
//    staged with vectorized float4 loads + f32->f16 convert.
//  - Generic path (K=22/41/82 entry GEMMs): guarded loads, register-batched.
// Param order assumes jax tree_leaves flattening (sorted dict keys):
//   7:dist.ff1 8:dist.ff2 9:dist.w0 10:emb 11:enc.ff1 12:enc.ff2 13:enc.w0
//   14:glocs.ff1 15:glocs.ff2 16:glocs.w0
//   17+12*i+{0:cx_wk 1:cx_wo 2:cx_wq 3:cx_wv 4:ff1 5:ff2 6:sq_wk 7:sq_wo
//            8:sq_wq 9:sq_wv 10:wg 11:wl}
//   161:locs.ff1 162:locs.ff2 163:locs.w0
// Workspace requirement ~330 MB (glocs buffer 201 MB dominates).
// ============================================================================

#define USE_TDM 1   // this toolchain uses the 6-arg tensor_load_to_lds builtin

namespace {

constexpr int Bsz = 2, L = 256, P = 256, D = 768, H = 12, NLn = 12, FIRSTn = 6, NION = 12;
constexpr int DFF = 4 * D;      // 3072
constexpr int GD  = D / 2;      // 384
constexpr float NORMf = 3000.0f;
constexpr long BLr    = (long)Bsz * L;                 // 512 activation rows
constexpr long ROWS_G = (long)Bsz * L * P;             // 131072 glocs rows
constexpr long ROWS_K = (long)Bsz * L * 2 * NION;      // 12288 locs rows
constexpr int  CH = 4096;                              // row chunk for big MLPs

typedef __attribute__((ext_vector_type(16))) _Float16 v16h;
typedef __attribute__((ext_vector_type(8)))  _Float16 v8h;
typedef __attribute__((ext_vector_type(8)))  float    v8f;
typedef __attribute__((ext_vector_type(4)))  unsigned int u32x4;
typedef __attribute__((ext_vector_type(8)))  int i32x8;
typedef __attribute__((ext_vector_type(4)))  int i32x4;
union H16 { v16h v; v8h h[2]; };

constexpr int BM = 64, BN = 64, BK = 32, LDH = BK + 8;  // f16 LDS stride (halfs)
constexpr int LDAF = BK + 4;                            // f32 A LDS stride (TDM pads 4 dw/row)

// ---------------------------------------------------------------------------
// FAST GEMM (M%64==0, N%64==0, K%32==0):
//   C = alpha * A @ B(^T) [+ Res] [ReLU], batched via z -> (b = z/Hdiv, h = z%Hdiv)
// A tile: TDM DMA (f32, padded rows), double-buffered; B tile: float4 loads.
// ---------------------------------------------------------------------------
__global__ __launch_bounds__(128)
void gemm_wmma_fast(const float* __restrict__ A, long ldA,
                    const float* __restrict__ Bm, long ldB,
                    float* __restrict__ C, long ldC,
                    const float* __restrict__ Res,
                    int M, int N, int K, float alpha,
                    int transB, int relu, int Hdiv,
                    long aSB, long aSH, long bSB, long bSH,
                    long cSB, long cSH, long rSB, long rSH)
{
  __shared__ __align__(16) float    AsF[2][BM * LDAF];
  __shared__ __align__(16) _Float16 Bs[2][BN * LDH];

  int z  = blockIdx.z;
  int zb = z / Hdiv, zh = z % Hdiv;
  A  += zb * aSB + zh * aSH;
  Bm += zb * bSB + zh * bSH;
  C  += zb * cSB + zh * cSH;
  if (Res) Res += zb * rSB + zh * rSH;

  int bm = blockIdx.y * BM;
  int bn = blockIdx.x * BN;
  int tid  = threadIdx.x;
  int lane = tid & 31;
  int wave = tid >> 5;
  int wm = (wave >> 1) * 32;
  int wn = (wave & 1) * 32;
  int nk = K / BK;

  // ---- B tile staging: vectorized float4, f32->f16 convert ----
  auto stageB = [&](int buf, int k0) {
    if (transB) {                       // B is NxK row-major: contiguous along K
#pragma unroll
      for (int it = 0; it < 4; ++it) {
        int slot = tid + it * 128;      // 512 float4 slots
        int n = slot >> 3, kk = (slot & 7) * 4;
        const float* p = &Bm[(long)(bn + n) * ldB + k0 + kk];
        float4 v = *(const float4*)p;
        if (k0 + 2 * BK < K) __builtin_prefetch(p + 2 * BK, 0, 0);
        _Float16* q = &Bs[buf][n * LDH + kk];
        q[0] = (_Float16)v.x; q[1] = (_Float16)v.y;
        q[2] = (_Float16)v.z; q[3] = (_Float16)v.w;
      }
    } else {                            // B is KxN: contiguous along N, scatter into Bs[n][k]
#pragma unroll
      for (int it = 0; it < 4; ++it) {
        int slot = tid + it * 128;
        int kk = slot >> 4, n4 = (slot & 15) * 4;
        const float* p = &Bm[(long)(k0 + kk) * ldB + bn + n4];
        float4 v = *(const float4*)p;
        if (k0 + 2 * BK < K) __builtin_prefetch(p + (long)2 * BK * ldB, 0, 0);
        Bs[buf][(n4 + 0) * LDH + kk] = (_Float16)v.x;
        Bs[buf][(n4 + 1) * LDH + kk] = (_Float16)v.y;
        Bs[buf][(n4 + 2) * LDH + kk] = (_Float16)v.z;
        Bs[buf][(n4 + 3) * LDH + kk] = (_Float16)v.w;
      }
    }
  };

  // ---- A tile -> LDS ----
#if USE_TDM
  // Tensor Data Mover: one instruction copies the 64x32 f32 tile, padding
  // 4 DWORDs after every 32-DWORD row (LDS row stride 144B). Issued by wave 0
  // only (EXEC-independent, once per wave); completion via s_wait_tensorcnt.
  auto tdmA = [&](int buf, int k0) {
    if (wave != 0) return;
    unsigned long long ga = (unsigned long long)(const void*)(A + (long)(bm)*ldA + k0);
    unsigned lds = (unsigned)(unsigned long long)(const void*)&AsF[buf][0];
    u32x4 g0;
    g0[0] = 1u;                                            // count=1, user mode
    g0[1] = lds;                                           // lds_addr
    g0[2] = (unsigned)(ga & 0xFFFFFFFFull);                // global_addr lo
    g0[3] = (unsigned)((ga >> 32) & 0x01FFFFFFull) | 0x80000000u; // hi | type=2
    i32x8 g1;
    g1[0] = (int)((2u << 16) | (1u << 20) | (4u << 22) | (3u << 25));
             // data_size=4B, pad_enable, pad_interval=32dw, pad_amount=4dw
    g1[1] = 0;                                             // td0 = 2^30 (low bits 0)
    g1[2] = 0x4000;                                        // td0 hi | td1 lo
    g1[3] = (int)(0x4000u | (32u << 16));                  // td1 hi | tile_dim0=32
    g1[4] = 64;                                            // tile_dim1=64 rows
    g1[5] = (int)(unsigned)ldA;                            // dim0 stride (elems)
    g1[6] = 0; g1[7] = 0;
    i32x4 zz = {0, 0, 0, 0};
    i32x8 z8 = {0, 0, 0, 0, 0, 0, 0, 0};
    __builtin_amdgcn_tensor_load_to_lds(g0, g1, zz, zz, z8, 0);
  };
#else
  auto tdmA = [&](int buf, int k0) {
#pragma unroll
    for (int it = 0; it < 4; ++it) {
      int slot = tid + it * 128;
      int r = slot >> 3, kk = (slot & 7) * 4;
      float4 v = *(const float4*)&A[(long)(bm + r) * ldA + k0 + kk];
      *(float4*)&AsF[buf][r * LDAF + kk] = v;
    }
  };
#endif

  v8f acc[2][2] = {};

  stageB(0, 0);
  tdmA(0, 0);
  for (int kt = 0; kt < nk; ++kt) {
    int cur = kt & 1, nxt = cur ^ 1;
#if USE_TDM
    if (wave == 0) __builtin_amdgcn_s_wait_tensorcnt(0);
#endif
    __syncthreads();                       // A[cur] + B[cur] ready; prior reads done
    if (kt + 1 < nk) {                     // overlap next tile with compute
      tdmA(nxt, (kt + 1) * BK);
      stageB(nxt, (kt + 1) * BK);
    }
    // A fragment from f32 LDS (ISA 7.12.2: lane=M row; K halves {kb..kb+7, kb+16..kb+23})
    v16h afrag[2], bfrag[2];
    int mrow = lane & 15;
    int kbA  = (lane >> 4) ? 8 : 0;
#pragma unroll
    for (int ti = 0; ti < 2; ++ti) {
      const float* p = &AsF[cur][(wm + 16 * ti + mrow) * LDAF + kbA];
      float4 a0 = *(const float4*)(p);
      float4 a1 = *(const float4*)(p + 4);
      float4 a2 = *(const float4*)(p + 16);
      float4 a3 = *(const float4*)(p + 20);
      H16 u;
      u.h[0][0] = (_Float16)a0.x; u.h[0][1] = (_Float16)a0.y;
      u.h[0][2] = (_Float16)a0.z; u.h[0][3] = (_Float16)a0.w;
      u.h[0][4] = (_Float16)a1.x; u.h[0][5] = (_Float16)a1.y;
      u.h[0][6] = (_Float16)a1.z; u.h[0][7] = (_Float16)a1.w;
      u.h[1][0] = (_Float16)a2.x; u.h[1][1] = (_Float16)a2.y;
      u.h[1][2] = (_Float16)a2.z; u.h[1][3] = (_Float16)a2.w;
      u.h[1][4] = (_Float16)a3.x; u.h[1][5] = (_Float16)a3.y;
      u.h[1][6] = (_Float16)a3.z; u.h[1][7] = (_Float16)a3.w;
      afrag[ti] = u.v;
    }
    int kbB = (lane >> 4) ? 8 : 0;
#pragma unroll
    for (int tj = 0; tj < 2; ++tj) {
      const _Float16* p = &Bs[cur][(wn + 16 * tj + (lane & 15)) * LDH + kbB];
      H16 u;
      u.h[0] = *(const v8h*)(p);
      u.h[1] = *(const v8h*)(p + 16);
      bfrag[tj] = u.v;
    }
#pragma unroll
    for (int ti = 0; ti < 2; ++ti)
#pragma unroll
      for (int tj = 0; tj < 2; ++tj)
        acc[ti][tj] = __builtin_amdgcn_wmma_f32_16x16x32_f16(
            false, afrag[ti], false, bfrag[tj], (short)0, acc[ti][tj],
            false, false);
  }

  int cn  = lane & 15;
  int rhi = (lane >> 4) * 8;
#pragma unroll
  for (int ti = 0; ti < 2; ++ti)
#pragma unroll
    for (int tj = 0; tj < 2; ++tj)
#pragma unroll
      for (int r = 0; r < 8; ++r) {
        int gm = bm + wm + 16 * ti + rhi + r;
        int gn = bn + wn + 16 * tj + cn;
        float v = acc[ti][tj][r] * alpha;
        if (Res) v += Res[(long)gm * ldC + gn];
        if (relu) v = fmaxf(v, 0.f);
        C[(long)gm * ldC + gn] = v;
      }
}

// ---------------------------------------------------------------------------
// GENERIC GEMM (unaligned dims: entry MLPs with K=22/41/82).
// Guarded loads batched into registers before LDS stores.
// ---------------------------------------------------------------------------
__global__ __launch_bounds__(128)
void gemm_wmma_generic(const float* __restrict__ A, long ldA,
                       const float* __restrict__ Bm, long ldB,
                       float* __restrict__ C, long ldC,
                       const float* __restrict__ Res,
                       int M, int N, int K, float alpha,
                       int transB, int relu, int Hdiv,
                       long aSB, long aSH, long bSB, long bSH,
                       long cSB, long cSH, long rSB, long rSH)
{
  __shared__ __align__(16) _Float16 As[BM * LDH];
  __shared__ __align__(16) _Float16 Bs[BN * LDH];

  int z  = blockIdx.z;
  int zb = z / Hdiv, zh = z % Hdiv;
  A  += zb * aSB + zh * aSH;
  Bm += zb * bSB + zh * bSH;
  C  += zb * cSB + zh * cSH;
  if (Res) Res += zb * rSB + zh * rSH;

  int bm = blockIdx.y * BM;
  int bn = blockIdx.x * BN;
  int tid  = threadIdx.x;
  int lane = tid & 31;
  int wave = tid >> 5;
  int wm = (wave >> 1) * 32;
  int wn = (wave & 1) * 32;

  v8f acc[2][2] = {};

  for (int k0 = 0; k0 < K; k0 += BK) {
    float av[16], bv[16];
#pragma unroll
    for (int it = 0; it < 16; ++it) {
      int idx = tid + it * 128;
      int r = idx >> 5, kk = idx & 31;
      int gm = bm + r, gk = k0 + kk;
      av[it] = (gm < M && gk < K) ? A[(long)gm * ldA + gk] : 0.f;
    }
#pragma unroll
    for (int it = 0; it < 16; ++it) {
      int idx = tid + it * 128;
      int n = idx >> 5, kk = idx & 31;
      int gn = bn + n, gk = k0 + kk;
      float v = 0.f;
      if (gn < N && gk < K)
        v = transB ? Bm[(long)gn * ldB + gk] : Bm[(long)gk * ldB + gn];
      bv[it] = v;
    }
#pragma unroll
    for (int it = 0; it < 16; ++it) {
      int idx = tid + it * 128;
      As[(idx >> 5) * LDH + (idx & 31)] = (_Float16)av[it];
    }
#pragma unroll
    for (int it = 0; it < 16; ++it) {
      int idx = tid + it * 128;
      Bs[(idx >> 5) * LDH + (idx & 31)] = (_Float16)bv[it];
    }
    __syncthreads();

    v16h afrag[2], bfrag[2];
    int mrow = lane & 15;
    int kbA  = (lane >> 4) ? 8 : 0;
#pragma unroll
    for (int ti = 0; ti < 2; ++ti) {
      const _Float16* p = &As[(wm + 16 * ti + mrow) * LDH];
      H16 u;
      u.h[0] = *(const v8h*)(p + kbA);
      u.h[1] = *(const v8h*)(p + kbA + 16);
      afrag[ti] = u.v;
    }
    int kbB = (lane >> 4) ? 8 : 0;
#pragma unroll
    for (int tj = 0; tj < 2; ++tj) {
      const _Float16* p = &Bs[(wn + 16 * tj + (lane & 15)) * LDH + kbB];
      H16 u;
      u.h[0] = *(const v8h*)(p);
      u.h[1] = *(const v8h*)(p + 16);
      bfrag[tj] = u.v;
    }
#pragma unroll
    for (int ti = 0; ti < 2; ++ti)
#pragma unroll
      for (int tj = 0; tj < 2; ++tj)
        acc[ti][tj] = __builtin_amdgcn_wmma_f32_16x16x32_f16(
            false, afrag[ti], false, bfrag[tj], (short)0, acc[ti][tj],
            false, false);
    __syncthreads();
  }

  int cn  = lane & 15;
  int rhi = (lane >> 4) * 8;
#pragma unroll
  for (int ti = 0; ti < 2; ++ti)
#pragma unroll
    for (int tj = 0; tj < 2; ++tj)
#pragma unroll
      for (int r = 0; r < 8; ++r) {
        int gm = bm + wm + 16 * ti + rhi + r;
        int gn = bn + wn + 16 * tj + cn;
        if (gm < M && gn < N) {
          float v = acc[ti][tj][r] * alpha;
          if (Res) v += Res[(long)gm * ldC + gn];
          if (relu) v = fmaxf(v, 0.f);
          C[(long)gm * ldC + gn] = v;
        }
      }
}

// ---------------------------------------------------------------------------
// LayerNorm rows (biased variance, eps 1e-5)
// ---------------------------------------------------------------------------
__global__ void ln_kernel(const float* __restrict__ src, float* __restrict__ dst, int n)
{
  __shared__ float red[256];
  long row = blockIdx.x;
  const float* s = src + row * (long)n;
  float* d = dst + row * (long)n;
  float sum = 0.f, sq = 0.f;
  for (int i = threadIdx.x; i < n; i += 256) { float v = s[i]; sum += v; sq += v * v; }
  red[threadIdx.x] = sum; __syncthreads();
  for (int o = 128; o; o >>= 1) { if (threadIdx.x < o) red[threadIdx.x] += red[threadIdx.x + o]; __syncthreads(); }
  float mean = red[0] / n; __syncthreads();
  red[threadIdx.x] = sq; __syncthreads();
  for (int o = 128; o; o >>= 1) { if (threadIdx.x < o) red[threadIdx.x] += red[threadIdx.x + o]; __syncthreads(); }
  float var = red[0] / n - mean * mean;
  float inv = rsqrtf(var + 1e-5f);
  for (int i = threadIdx.x; i < n; i += 256) d[i] = (s[i] - mean) * inv;
}

// ---------------------------------------------------------------------------
// Softmax (self: causal & aa>0 key mask; cross: key < p_len). Row len = 256.
// ---------------------------------------------------------------------------
__global__ void softmax_self_kernel(float* __restrict__ s, const int* __restrict__ aa)
{
  __shared__ float red[256];
  long row = blockIdx.x;                 // (b*H + h)*L + i
  int i = (int)(row % L);
  int b = (int)(row / ((long)H * L));
  float* sr = s + row * (long)L;
  int j = threadIdx.x;
  bool valid = (aa[b * L + j] > 0) && (j <= i);
  float v = valid ? sr[j] : -1e9f;
  red[j] = v; __syncthreads();
  for (int o = 128; o; o >>= 1) { if (j < o) red[j] = fmaxf(red[j], red[j + o]); __syncthreads(); }
  float mx = red[0]; __syncthreads();
  float e = __expf(v - mx);
  red[j] = e; __syncthreads();
  for (int o = 128; o; o >>= 1) { if (j < o) red[j] += red[j + o]; __syncthreads(); }
  sr[j] = e / red[0];
}

__global__ void softmax_cross_kernel(float* __restrict__ s, const int* __restrict__ p_len)
{
  __shared__ float red[256];
  long row = blockIdx.x;                 // (b*H + h)*L + i
  int b = (int)(row / ((long)H * L));
  float* sr = s + row * (long)P;
  int j = threadIdx.x;
  bool valid = j < p_len[b];
  float v = valid ? sr[j] : -1e9f;
  red[j] = v; __syncthreads();
  for (int o = 128; o; o >>= 1) { if (j < o) red[j] = fmaxf(red[j], red[j + o]); __syncthreads(); }
  float mx = red[0]; __syncthreads();
  float e = __expf(v - mx);
  red[j] = e; __syncthreads();
  for (int o = 128; o; o >>= 1) { if (j < o) red[j] += red[j + o]; __syncthreads(); }
  sr[j] = e / red[0];
}

// ---------------------------------------------------------------------------
// bias[b,h,l,p] = sum_f glocs[(b,l,p),f] * wg[f,h]    (wave per glocs row)
// ---------------------------------------------------------------------------
__global__ void bias_g_kernel(const float* __restrict__ glocs,
                              const float* __restrict__ wg,
                              float* __restrict__ bias)
{
  long r = (long)blockIdx.x * 8 + (threadIdx.x >> 5);
  if (r >= ROWS_G) return;
  int lane = threadIdx.x & 31;
  const float* g = glocs + r * GD;
  float acc[H] = {};
  for (int f = lane; f < GD; f += 32) {
    float v = g[f];
#pragma unroll
    for (int h = 0; h < H; ++h) acc[h] += v * wg[f * H + h];
  }
  int b = (int)(r / ((long)L * P));
  int l = (int)((r / P) % L);
  int p = (int)(r % P);
#pragma unroll
  for (int h = 0; h < H; ++h) {
    float sv = acc[h];
    for (int o = 16; o; o >>= 1) sv += __shfl_xor(sv, o, 32);
    if (lane == 0) bias[(((long)b * H + h) * L + l) * P + p] = sv;
  }
}

// lv = locs @ wl, scatter-add into bias at knn indices (owner = (b,l,h))
__global__ void lv_scatter_kernel(const float* __restrict__ locs,
                                  const float* __restrict__ wl,
                                  const int* __restrict__ idx24,
                                  float* __restrict__ bias)
{
  int t = blockIdx.x * blockDim.x + threadIdx.x;   // (b*L+l)*H + h
  if (t >= Bsz * L * H) return;
  int h = t % H; int bl = t / H;
  int b = bl / L, l = bl % L;
  for (int k = 0; k < 2 * NION; ++k) {
    const float* row = locs + ((long)bl * (2 * NION) + k) * D;
    float s = 0.f;
    for (int d = 0; d < D; ++d) s += row[d] * wl[d * H + h];
    int p = idx24[bl * (2 * NION) + k];
    bias[(((long)b * H + h) * L + l) * P + p] += s;
  }
}

// ---------------------------------------------------------------------------
// Feature / init kernels
// ---------------------------------------------------------------------------
__device__ __forceinline__ float embed_feat(float x, int j)
{
  if (j < 20)  return __sinf(x * (float)(1u << j));
  if (j == 20) return x;
  return __cosf(x * (float)(1u << (j - 21)));
}

__global__ void init_x_kernel(const int* __restrict__ aa, const float* __restrict__ emb,
                              float* __restrict__ x)
{
  long t = (long)blockIdx.x * 256 + threadIdx.x;
  if (t >= BLr * D) return;
  int d = (int)(t % D);
  long bl = t / D;
  int l = (int)(bl % L);
  float ang = (float)l * __expf(-(2.f * (float)(d >> 1) / (float)D) * logf(10000.f));
  float pe = (d & 1) ? __cosf(ang) : __sinf(ang);
  x[t] = emb[(long)aa[bl] * D + d] + pe;
}

__global__ void knn_kernel(const float* __restrict__ c_ion, const float* __restrict__ mz,
                           const int* __restrict__ p_len, const unsigned char* __restrict__ aa_mask,
                           float* __restrict__ knnd, int* __restrict__ knni)
{
  int t = blockIdx.x * blockDim.x + threadIdx.x;   // b*L*NION + li
  if (t >= Bsz * L * NION) return;
  int b = t / (L * NION);
  float q = c_ion[t] * (1.f / NORMf);
  int pl = p_len[b];
  float b0 = 3.4e38f, b1 = 3.4e38f; int i0 = 0, i1 = 0;
  for (int p = 0; p < P; ++p) {
    float d = q - mz[b * P + p] * (1.f / NORMf);
    float d2 = (p < pl) ? d * d : 3.4e38f;
    if (d2 < b0)      { b1 = b0; i1 = i0; b0 = d2; i0 = p; }
    else if (d2 < b1) { b1 = d2; i1 = p; }
  }
  int bl = t / NION;
  float m = aa_mask[bl] ? 1.f : 0.f;
  long o = (long)t * 2;
  knnd[o]     = -sqrtf(b0) * m;
  knnd[o + 1] = -sqrtf(b1) * m;
  knni[o] = i0; knni[o + 1] = i1;
}

__global__ void embed_knn_kernel(const float* __restrict__ knnd, float* __restrict__ knnf)
{
  long t = (long)blockIdx.x * 256 + threadIdx.x;
  if (t >= ROWS_K * 41) return;
  long r = t / 41; int j = (int)(t % 41);
  knnf[t] = embed_feat(knnd[r], j);
}

__global__ void embed_mz_kernel(const float* __restrict__ mz, float* __restrict__ mzf)
{
  long t = (long)blockIdx.x * 256 + threadIdx.x;
  if (t >= (long)Bsz * P * 41) return;
  long r = t / 41; int j = (int)(t % 41);
  mzf[t] = embed_feat(mz[r] * (1.f / NORMf), j);
}

// gl features for one 4096-row chunk: row=(b,l,p), 82 feats from hm0/hm1 - mzn
__global__ void embed_gl_kernel(const float* __restrict__ h_mass, const float* __restrict__ mz,
                                float* __restrict__ glf, long r0)
{
  long t = (long)blockIdx.x * 256 + threadIdx.x;
  if (t >= (long)CH * 82) return;
  long rr = t / 82; int f = (int)(t % 82);
  long row = r0 + rr;
  if (row >= ROWS_G) return;
  int b = (int)(row / ((long)L * P));
  int l = (int)((row / P) % L);
  int p = (int)(row % P);
  float hm = h_mass[((long)b * L + l) * 2 + (f >= 41 ? 1 : 0)] * (1.f / NORMf);
  float x = hm - mz[(long)b * P + p] * (1.f / NORMf);
  glf[t] = embed_feat(x, f >= 41 ? f - 41 : f);
}

__global__ void add_kernel(const float* __restrict__ a, const float* __restrict__ b,
                           float* __restrict__ c, long n)
{
  long t = (long)blockIdx.x * 256 + threadIdx.x;
  if (t < n) c[t] = a[t] + b[t];
}

// ---------------------------------------------------------------------------
// Host-side helpers
// ---------------------------------------------------------------------------
inline unsigned cdiv(long a, long b) { return (unsigned)((a + b - 1) / b); }

void gemm(hipStream_t st, const float* A, long ldA, const float* B, long ldB,
          float* C, long ldC, const float* Res, int M, int N, int K,
          float alpha, bool transB, bool relu,
          int batches = 1, int Hdiv = 1,
          long aSB = 0, long aSH = 0, long bSB = 0, long bSH = 0,
          long cSB = 0, long cSH = 0, long rSB = 0, long rSH = 0)
{
  dim3 g(cdiv(N, BN), cdiv(M, BM), batches);
  bool aligned = (M % BM == 0) && (N % BN == 0) && (K % BK == 0);
  if (aligned)
    gemm_wmma_fast<<<g, 128, 0, st>>>(A, ldA, B, ldB, C, ldC, Res, M, N, K, alpha,
                                      transB ? 1 : 0, relu ? 1 : 0, Hdiv,
                                      aSB, aSH, bSB, bSH, cSB, cSH, rSB, rSH);
  else
    gemm_wmma_generic<<<g, 128, 0, st>>>(A, ldA, B, ldB, C, ldC, Res, M, N, K, alpha,
                                         transB ? 1 : 0, relu ? 1 : 0, Hdiv,
                                         aSB, aSH, bSB, bSH, cSB, cSH, rSB, rSH);
}

// mlp_block(x) = pff(relu(x @ w0)) ; pff(h)=ln(h + relu(h@ff1)@ff2)
void mlp_block_rows(hipStream_t st, const float* in, int rows, int din,
                    const float* w0, const float* ff1, const float* ff2,
                    int d, int dff, float* out, float* h0, float* t)
{
  gemm(st, in, din, w0, d, h0, d, nullptr, rows, d, din, 1.f, false, true);
  gemm(st, h0, d, ff1, dff, t, dff, nullptr, rows, dff, d, 1.f, false, true);
  gemm(st, t, dff, ff2, d, out, d, /*Res=*/h0, rows, d, dff, 1.f, false, false);
  ln_kernel<<<rows, 256, 0, st>>>(out, out, d);
}

} // namespace

// ============================================================================
extern "C" void kernel_launch(void* const* d_in, const int* in_sizes, int n_in,
                              void* d_out, int out_size, void* d_ws, size_t ws_size,
                              hipStream_t stream)
{
  (void)in_sizes; (void)n_in; (void)out_size; (void)ws_size;
  hipStream_t st = stream;

  const int*   aa       = (const int*)  d_in[0];
  const float* h_mass   = (const float*)d_in[1];
  const float* mz       = (const float*)d_in[2];
  const float* c_ion    = (const float*)d_in[3];
  const float* distance = (const float*)d_in[4];
  const unsigned char* aa_mask = (const unsigned char*)d_in[5];
  const int*   p_len    = (const int*)  d_in[6];

  const float* dist_ff1 = (const float*)d_in[7];
  const float* dist_ff2 = (const float*)d_in[8];
  const float* dist_w0  = (const float*)d_in[9];
  const float* embw     = (const float*)d_in[10];
  const float* enc_ff1  = (const float*)d_in[11];
  const float* enc_ff2  = (const float*)d_in[12];
  const float* enc_w0   = (const float*)d_in[13];
  const float* gl_ff1   = (const float*)d_in[14];
  const float* gl_ff2   = (const float*)d_in[15];
  const float* gl_w0    = (const float*)d_in[16];
  auto LP = [&](int i, int j) { return (const float*)d_in[17 + i * 12 + j]; };
  const float* locs_ff1 = (const float*)d_in[17 + 12 * NLn + 0];
  const float* locs_ff2 = (const float*)d_in[17 + 12 * NLn + 1];
  const float* locs_w0  = (const float*)d_in[17 + 12 * NLn + 2];

  // ---- workspace layout (f32, 256B aligned) ----
  float* W = (float*)d_ws;
  long off = 0;
  auto alloc = [&](long n) { float* p = W + off; off += (n + 63) & ~63L; return p; };
  float* x     = alloc(BLr * D);
  float* xr    = alloc(BLr * D);
  float* hbuf  = alloc(BLr * D);
  float* qb    = alloc(BLr * D);
  float* kb    = alloc(BLr * D);
  float* vb    = alloc(BLr * D);
  float* ob    = alloc(BLr * D);
  float* distf = alloc(BLr * D);
  float* massb = alloc(BLr * D);
  float* sb    = alloc((long)Bsz * H * L * P);
  float* biasb = alloc((long)Bsz * H * L * P);
  float* tb    = alloc((long)CH * DFF);     // big FFN intermediate (chunked)
  float* c0    = alloc((long)CH * D);       // h0 chunks
  float* glocs = alloc(ROWS_G * GD);        // 201 MB persistent
  float* locsb = alloc(ROWS_K * D);
  float* knnf  = alloc(ROWS_K * 41);
  float* knnd  = alloc(ROWS_K);
  int*   knni  = (int*)alloc(ROWS_K);
  float* mzf   = alloc((long)Bsz * P * 41);
  float* glf   = alloc((long)CH * 82);

  // ---- precompute ----
  init_x_kernel<<<cdiv(BLr * D, 256), 256, 0, st>>>(aa, embw, x);
  knn_kernel<<<cdiv(Bsz * L * NION, 256), 256, 0, st>>>(c_ion, mz, p_len, aa_mask, knnd, knni);
  embed_knn_kernel<<<cdiv(ROWS_K * 41, 256), 256, 0, st>>>(knnd, knnf);
  embed_mz_kernel<<<cdiv((long)Bsz * P * 41, 256), 256, 0, st>>>(mz, mzf);

  // dist_feat = mlp_block(distance)     rows=512, K=22
  mlp_block_rows(st, distance, (int)BLr, 22, dist_w0, dist_ff1, dist_ff2, D, DFF, distf, c0, tb);
  // mass = mlp_block(embeder(mzn))      rows=512, K=41
  mlp_block_rows(st, mzf, (int)BLr, 41, enc_w0, enc_ff1, enc_ff2, D, DFF, massb, c0, tb);
  // locs = mlp_block(knn_feat)          rows=12288, K=41 (chunked)
  for (long r0 = 0; r0 < ROWS_K; r0 += CH)
    mlp_block_rows(st, knnf + r0 * 41, CH, 41, locs_w0, locs_ff1, locs_ff2, D, DFF,
                   locsb + r0 * D, c0, tb);
  // glocs = mlp_block(gl)               rows=131072, K=82 (gl computed per chunk)
  for (long r0 = 0; r0 < ROWS_G; r0 += CH) {
    embed_gl_kernel<<<cdiv((long)CH * 82, 256), 256, 0, st>>>(h_mass, mz, glf, r0);
    mlp_block_rows(st, glf, CH, 82, gl_w0, gl_ff1, gl_ff2, GD, 4 * GD,
                   glocs + r0 * GD, c0, tb);
  }

  // ---- transformer layers ----
  const long LL = (long)L * L, LPn = (long)L * P, LD = (long)L * D, PD = (long)P * D;
  for (int i = 0; i < NLn; ++i) {
    bool local = i < FIRSTn;
    // self-attention (h = x + dist_feat for local layers)
    const float* h = x;
    if (local) {
      add_kernel<<<cdiv(BLr * D, 256), 256, 0, st>>>(x, distf, hbuf, BLr * D);
      h = hbuf;
    }
    gemm(st, h, D, LP(i, 8), D, qb, D, nullptr, (int)BLr, D, D, 1.f, false, false); // sq_wq
    gemm(st, h, D, LP(i, 6), D, kb, D, nullptr, (int)BLr, D, D, 1.f, false, false); // sq_wk
    gemm(st, h, D, LP(i, 9), D, vb, D, nullptr, (int)BLr, D, D, 1.f, false, false); // sq_wv
    // scores = q @ k^T / 8 per (b,h)
    gemm(st, qb, D, kb, D, sb, L, nullptr, L, L, D / H, 0.125f, true, false,
         Bsz * H, H, LD, 64, LD, 64, (long)H * LL, LL, 0, 0);
    softmax_self_kernel<<<Bsz * H * L, 256, 0, st>>>(sb, aa);
    // o = a @ v per (b,h)
    gemm(st, sb, L, vb, D, ob, D, nullptr, L, D / H, L, 1.f, false, false,
         Bsz * H, H, (long)H * LL, LL, LD, 64, LD, 64);
    gemm(st, ob, D, LP(i, 7), D, xr, D, /*Res=*/x, (int)BLr, D, D, 1.f, false, false); // sq_wo
    ln_kernel<<<(int)BLr, 256, 0, st>>>(xr, x, D);

    // cross-attention bias: glocs @ wg (+ local scatter of locs @ wl)
    bias_g_kernel<<<cdiv(ROWS_G, 8), 256, 0, st>>>(glocs, LP(i, 10), biasb);
    if (local)
      lv_scatter_kernel<<<cdiv(Bsz * L * H, 256), 256, 0, st>>>(locsb, LP(i, 11), knni, biasb);

    // cross-attention: q from x, k/v from mass
    gemm(st, x,     D, LP(i, 2), D, qb, D, nullptr, (int)BLr, D, D, 1.f, false, false); // cx_wq
    gemm(st, massb, D, LP(i, 0), D, kb, D, nullptr, (int)BLr, D, D, 1.f, false, false); // cx_wk
    gemm(st, massb, D, LP(i, 3), D, vb, D, nullptr, (int)BLr, D, D, 1.f, false, false); // cx_wv
    gemm(st, qb, D, kb, D, sb, P, /*Res=bias*/ biasb, L, P, D / H, 0.125f, true, false,
         Bsz * H, H, LD, 64, PD, 64, (long)H * LPn, LPn, (long)H * LPn, LPn);
    softmax_cross_kernel<<<Bsz * H * L, 256, 0, st>>>(sb, p_len);
    gemm(st, sb, P, vb, D, ob, D, nullptr, L, D / H, P, 1.f, false, false,
         Bsz * H, H, (long)H * LPn, LPn, PD, 64, LD, 64);
    gemm(st, ob, D, LP(i, 1), D, xr, D, /*Res=*/x, (int)BLr, D, D, 1.f, false, false); // cx_wo
    ln_kernel<<<(int)BLr, 256, 0, st>>>(xr, x, D);

    // feed-forward
    gemm(st, x, D, LP(i, 4), DFF, tb, DFF, nullptr, (int)BLr, DFF, D, 1.f, false, true);
    gemm(st, tb, DFF, LP(i, 5), D, xr, D, /*Res=*/x, (int)BLr, D, DFF, 1.f, false, false);
    float* dst = (i == NLn - 1) ? (float*)d_out : x;
    ln_kernel<<<(int)BLr, 256, 0, st>>>(xr, dst, D);
  }
}
